// RelationalMP_3324304687538
// MI455X (gfx1250) — matrix operand order repaired and checked by reading, not compiled, not run on hardware
//
#include <hip/hip_runtime.h>

typedef _Float16 v8h  __attribute__((ext_vector_type(8)));
typedef _Float16 v16h __attribute__((ext_vector_type(16)));
typedef float    v8f  __attribute__((ext_vector_type(8)));

#define HDIM 128
#define K1   256          // 2*H
#define TILE_E 16
#define EDGES_PER_BLOCK 512
#define NTHREADS 256

// ---------------------------------------------------------------------------
// Zero the output accumulator (harness poisons d_out with 0xAA).
// ---------------------------------------------------------------------------
__global__ __launch_bounds__(256) void relmp_zero(float* __restrict__ out, int n) {
  int i = (blockIdx.x * blockDim.x + threadIdx.x) * 4;
  if (i + 3 < n) {
    *(float4*)(out + i) = make_float4(0.f, 0.f, 0.f, 0.f);
  } else {
    for (int j = i; j < n; ++j) out[j] = 0.f;
  }
}

// ---------------------------------------------------------------------------
// Fused gather -> MLP(2 layer, relu) -> scatter-add, per edge type.
// blockIdx.y = edge type t, blockIdx.x = 512-edge chunk. 256 thr = 8 waves.
// Wave w computes output columns [16w, 16w+16) of both GEMMs via
// v_wmma_f32_16x16x32_f16 with LDS-staged f16 operands.
// ---------------------------------------------------------------------------
__global__ __launch_bounds__(NTHREADS)
void relmp_kernel(const float* __restrict__ node_states,  // [N][128]
                  const int*   __restrict__ adj,          // [T][E][2]
                  const float* __restrict__ W1g,          // [T][256][128]
                  const float* __restrict__ W2g,          // [T][128][128]
                  float* __restrict__ out,                // [N][128]
                  int E, int N)
{
  // Weights pre-swizzled into B-fragment order: 16 contiguous halfs per lane.
  __shared__ __align__(32) _Float16 sW1[K1 * HDIM];     // 64 KB
  __shared__ __align__(32) _Float16 sW2[HDIM * HDIM];   // 32 KB
  __shared__ __align__(32) _Float16 sX [TILE_E * K1];   //  8 KB, row-major
  __shared__ __align__(32) _Float16 sH [TILE_E * HDIM]; //  4 KB, row-major
  __shared__ int sTgt[TILE_E];

  const int t    = blockIdx.y;
  const int tid  = threadIdx.x;
  const int lane = tid & 31;
  const int wave = tid >> 5;

  // ---- stage + swizzle weights (f32 -> f16) into LDS, once per block ------
  // B-fragment layout (32x16, f16): lane l holds col N = l&15; half j in the
  // lane's 16-half run is K = kstep*32 + 16*(l>>4) + j.
  const float* W1t = W1g + (size_t)t * K1 * HDIM;
  for (int idx = tid; idx < K1 * HDIM; idx += NTHREADS) {
    int Kr = idx >> 7, Nc = idx & 127;
    int kstep = Kr >> 5, rem = Kr & 31;
    int hi = rem >> 4, j = rem & 15;
    int nblk = Nc >> 4, l = (hi << 4) | (Nc & 15);
    sW1[(((nblk << 3) + kstep) * 32 + l) * 16 + j] = (_Float16)W1t[idx];
  }
  const float* W2t = W2g + (size_t)t * HDIM * HDIM;
  for (int idx = tid; idx < HDIM * HDIM; idx += NTHREADS) {
    int Kr = idx >> 7, Nc = idx & 127;
    int kstep = Kr >> 5, rem = Kr & 31;
    int hi = rem >> 4, j = rem & 15;
    int nblk = Nc >> 4, l = (hi << 4) | (Nc & 15);
    sW2[(((nblk << 2) + kstep) * 32 + l) * 16 + j] = (_Float16)W2t[idx];
  }
  __syncthreads();

  const int e0   = blockIdx.x * EDGES_PER_BLOCK;
  const int eEnd = min(e0 + EDGES_PER_BLOCK, E);

  // A-fragment addressing (16x32 f16): lane row = lane&15; lanes 0-15 hold
  // K in {0..15}, lanes 16-31 hold K in {16..31}.
  const int arow = lane & 15;
  const int aoff = (lane >> 4) << 4;   // 0 or 16
  const int crow = (lane >> 4) << 3;   // C/D: vgpr r -> row r + crow
  const int ccol = lane & 15;

  for (int tb = e0; tb < eEnd; tb += TILE_E) {
    // ---- gather + convert the 16x256 x-tile (concat src/tgt states) ------
    {
      int el = tid >> 4;              // local edge 0..15
      int ch = tid & 15;              // 16-element chunk 0..15
      int e  = tb + el;
      if (e < E) {
        const size_t abase = ((size_t)t * E + e) * 2;
        int src = adj[abase + 0];
        int tgt = adj[abase + 1];
        if (ch == 0) {
          sTgt[el] = tgt;
          // prefetch next tile's adjacency through L2
          size_t nxt = ((size_t)t * E + min(e + TILE_E, E - 1)) * 2;
          __builtin_prefetch(adj + nxt, 0, 1);
        }
        const float* base = (ch < 8)
            ? node_states + (size_t)src * HDIM + (ch << 4)
            : node_states + (size_t)tgt * HDIM + ((ch - 8) << 4);
        const float4* p = (const float4*)base;
        float4 f0 = p[0], f1 = p[1], f2 = p[2], f3 = p[3];
        v8h h0, h1;
        h0[0]=(_Float16)f0.x; h0[1]=(_Float16)f0.y; h0[2]=(_Float16)f0.z; h0[3]=(_Float16)f0.w;
        h0[4]=(_Float16)f1.x; h0[5]=(_Float16)f1.y; h0[6]=(_Float16)f1.z; h0[7]=(_Float16)f1.w;
        h1[0]=(_Float16)f2.x; h1[1]=(_Float16)f2.y; h1[2]=(_Float16)f2.z; h1[3]=(_Float16)f2.w;
        h1[4]=(_Float16)f3.x; h1[5]=(_Float16)f3.y; h1[6]=(_Float16)f3.z; h1[7]=(_Float16)f3.w;
        *(v8h*)(sX + el * K1 + (ch << 4))     = h0;
        *(v8h*)(sX + el * K1 + (ch << 4) + 8) = h1;
      } else {
        v8h z = {};
        *(v8h*)(sX + el * K1 + (ch << 4))     = z;
        *(v8h*)(sX + el * K1 + (ch << 4) + 8) = z;
        if (ch == 0) sTgt[el] = 0;
      }
    }
    __syncthreads();

    // ---- layer 1: [16x256] x [256x16] per wave, K-steps of 32 ------------
    v8f acc = {};
    #pragma unroll
    for (int k = 0; k < 8; ++k) {
      v8h lo = *(const v8h*)(sX + arow * K1 + (k << 5) + aoff);
      v8h hi = *(const v8h*)(sX + arow * K1 + (k << 5) + aoff + 8);
      v16h a = __builtin_shufflevector(lo, hi, 0,1,2,3,4,5,6,7,8,9,10,11,12,13,14,15);
      v16h b = *(const v16h*)(sW1 + (((wave << 3) + k) * 32 + lane) * 16);
      acc = __builtin_amdgcn_wmma_f32_16x16x32_f16(false, a, false, b,
                                                   (short)0, acc, false, false);
    }
    // relu -> f16 h-tile in LDS
    #pragma unroll
    for (int r = 0; r < 8; ++r) {
      float v = acc[r];
      v = v > 0.f ? v : 0.f;
      sH[(crow + r) * HDIM + (wave << 4) + ccol] = (_Float16)v;
    }
    __syncthreads();

    // ---- layer 2: [16x128] x [128x16] per wave ---------------------------
    v8f acc2 = {};
    #pragma unroll
    for (int k = 0; k < 4; ++k) {
      v8h lo = *(const v8h*)(sH + arow * HDIM + (k << 5) + aoff);
      v8h hi = *(const v8h*)(sH + arow * HDIM + (k << 5) + aoff + 8);
      v16h a = __builtin_shufflevector(lo, hi, 0,1,2,3,4,5,6,7,8,9,10,11,12,13,14,15);
      v16h b = *(const v16h*)(sW2 + (((wave << 2) + k) * 32 + lane) * 16);
      acc2 = __builtin_amdgcn_wmma_f32_16x16x32_f16(false, a, false, b,
                                                    (short)0, acc2, false, false);
    }
    // relu + scatter-add into out[tgt] (L2-resident f32 atomics)
    {
      const int ocol = (wave << 4) + ccol;
      #pragma unroll
      for (int r = 0; r < 8; ++r) {
        int row = crow + r;
        if (tb + row < E) {
          float v = acc2[r];
          v = v > 0.f ? v : 0.f;
          int tgt = sTgt[row];
          __hip_atomic_fetch_add(out + (size_t)tgt * HDIM + ocol, v,
                                 __ATOMIC_RELAXED, __HIP_MEMORY_SCOPE_AGENT);
        }
      }
    }
    __syncthreads();  // protect sX/sH/sTgt before next tile's gather
  }
}

// ---------------------------------------------------------------------------
extern "C" void kernel_launch(void* const* d_in, const int* in_sizes, int n_in,
                              void* d_out, int out_size, void* d_ws, size_t ws_size,
                              hipStream_t stream) {
  const float* node_states = (const float*)d_in[0];
  const int*   adj         = (const int*)d_in[1];
  const float* W1          = (const float*)d_in[2];
  const float* W2          = (const float*)d_in[3];
  float*       out         = (float*)d_out;

  const int N = in_sizes[0] / HDIM;                 // 100000
  const int T = in_sizes[2] / (2 * HDIM * HDIM);    // 8
  const int E = in_sizes[1] / (T * 2);              // 150000

  // zero the accumulator
  int zb = (out_size / 4 + 255) / 256;
  relmp_zero<<<zb, 256, 0, stream>>>(out, out_size);

  dim3 grid((E + EDGES_PER_BLOCK - 1) / EDGES_PER_BLOCK, T);
  relmp_kernel<<<grid, NTHREADS, 0, stream>>>(node_states, adj, W1, W2, out, E, N);
}
